// AttentionModuleForQwen2_1537598292454
// MI455X (gfx1250) — matrix operand order repaired and checked by reading, not compiled, not run on hardware
//
#include <hip/hip_runtime.h>

// ---------------------------------------------------------------------------
// MI455X (gfx1250) hogwild-attention for Qwen2: W=2, QN=64, H=28, HKV=4,
// D=128, HID=3584, fragments {8192,4096,2048}. bf16 WMMA, f32 accumulate.
// K/V pre-converted to L2-resident bf16 caches (V pre-transposed), attention
// streams tiles via async global->LDS double buffering.
// ---------------------------------------------------------------------------

typedef __attribute__((ext_vector_type(16))) __bf16 v16bf;
typedef __attribute__((ext_vector_type(8)))  float  v8f;
typedef __attribute__((ext_vector_type(4)))  float  f4raw;
typedef __attribute__((ext_vector_type(4)))  int    v4i;

#define WMMA_BF16(a, b, c) \
  __builtin_amdgcn_wmma_f32_16x16x32_bf16(false, (a), false, (b), (short)0, (c), false, false)

#if defined(__has_builtin)
#if __has_builtin(__builtin_amdgcn_global_load_async_to_lds_b128)
#define HAS_ASYNC 1
#endif
#endif
#ifndef HAS_ASYNC
#define HAS_ASYNC 0
#endif

#if HAS_ASYNC
__device__ __forceinline__ void async_cp16(const void* g, void* l) {
  __builtin_amdgcn_global_load_async_to_lds_b128(
      (__attribute__((address_space(1))) v4i*)g,
      (__attribute__((address_space(3))) v4i*)l, 0, 0);
}
#endif
__device__ __forceinline__ void wait_async() {
#if HAS_ASYNC
#if __has_builtin(__builtin_amdgcn_s_wait_asynccnt)
  __builtin_amdgcn_s_wait_asynccnt(0);
#else
  asm volatile("s_wait_asynccnt 0x0" ::: "memory");
#endif
#endif
}

__device__ __forceinline__ unsigned short f2bf(float x) {
  unsigned u = __builtin_bit_cast(unsigned, x);
  u += 0x7FFFu + ((u >> 16) & 1u);           // round-to-nearest-even
  return (unsigned short)(u >> 16);
}
__device__ __forceinline__ unsigned pack2(float a, float b) {
  return (unsigned)f2bf(a) | ((unsigned)f2bf(b) << 16);
}

union ABFrag { v16bf v; f4raw f[2]; };

// A-matrix 16x32 bf16 fragment (ISA 7.12.2): lane m=l&15, hi=l>>4.
__device__ __forceinline__ v16bf load_frag_a(const unsigned short* row, int kbase, int hi) {
  ABFrag u;
  u.f[0] = *(const f4raw*)(row + kbase + hi * 8);
  u.f[1] = *(const f4raw*)(row + kbase + 16 + hi * 8);
  return u.v;
}
// B-matrix 32x16 bf16 fragment: lane n=l&15, hi=l>>4; k = hi*8 + 0..7, 16 + hi*8 + 0..7.
__device__ __forceinline__ v16bf load_frag_b(const unsigned short* col, int hi) {
  ABFrag u;
  u.f[0] = *(const f4raw*)(col + hi * 8);
  u.f[1] = *(const f4raw*)(col + 16 + hi * 8);
  return u.v;
}
__device__ __forceinline__ v8f vzero8() {
  v8f r;
  for (int i = 0; i < 8; ++i) r[i] = 0.f;
  return r;
}

// ---------------------------------------------------------------------------
// K cache: flat fp32 -> bf16 (layout preserved [kv][L][D]).
// ---------------------------------------------------------------------------
__global__ __launch_bounds__(256) void convert_flat(const float* __restrict__ src,
                                                    unsigned short* __restrict__ dst, int n4) {
  int i = blockIdx.x * 256 + threadIdx.x;
  if (i >= n4) return;
  f4raw v = *(const f4raw*)(src + (size_t)i * 4);
  unsigned* o = (unsigned*)(dst + (size_t)i * 4);
  o[0] = pack2(v[0], v[1]);
  o[1] = pack2(v[2], v[3]);
}

// ---------------------------------------------------------------------------
// V cache: [kv][L][D] fp32 -> transposed bf16 [kv][D][L] (coalesced stores).
// ---------------------------------------------------------------------------
__global__ __launch_bounds__(256) void transpose_v(const float* __restrict__ v,
                                                   unsigned short* __restrict__ vt, int L) {
  int L4 = L >> 2;
  int idx = blockIdx.x * 256 + threadIdx.x;          // over 4*128*L4
  if (idx >= 4 * 128 * L4) return;
  int k4 = idx % L4;
  int rest = idx / L4;                                // kv*128 + d
  int d = rest & 127, kv = rest >> 7;
  (void)d; (void)kv;
  const float* base = v + ((size_t)(rest >> 7) * L + (size_t)k4 * 4) * 128 + (rest & 127);
  unsigned short r0 = f2bf(base[0]);
  unsigned short r1 = f2bf(base[128]);
  unsigned short r2 = f2bf(base[256]);
  unsigned short r3 = f2bf(base[384]);
  unsigned* o = (unsigned*)(vt + (size_t)rest * L + (size_t)k4 * 4);
  o[0] = (unsigned)r0 | ((unsigned)r1 << 16);
  o[1] = (unsigned)r2 | ((unsigned)r3 << 16);
}

// ---------------------------------------------------------------------------
// C[M,N] fp32 = A[M,K] * B[N,K]^T (+bias), bf16 WMMA, inline fp32->bf16 stage.
// grid = (M/64, N/64), block = 128 (4 waves).
// ---------------------------------------------------------------------------
__global__ __launch_bounds__(128) void gemm_bf16(const float* __restrict__ A,
                                                 const float* __restrict__ B,
                                                 const float* __restrict__ bias,
                                                 float* __restrict__ C,
                                                 int M, int N, int K) {
  __shared__ unsigned short Al[64 * 72];
  __shared__ unsigned short Bl[64 * 72];
  const int t = threadIdx.x, wave = t >> 5, lane = t & 31;
  const int nl = lane & 15, hi = lane >> 4;
  const int m0 = blockIdx.x * 64, n0 = blockIdx.y * 64;

  v8f acc[4];
  for (int i = 0; i < 4; ++i) acc[i] = vzero8();

  for (int k0 = 0; k0 < K; k0 += 64) {
    __syncthreads();
    for (int ii = t; ii < 1024; ii += 128) {
      int r = ii >> 4, kq = (ii & 15) << 2;
      f4raw av = *(const f4raw*)(A + (size_t)(m0 + r) * K + k0 + kq);
      unsigned* ad = (unsigned*)(Al + r * 72 + kq);
      ad[0] = pack2(av[0], av[1]);
      ad[1] = pack2(av[2], av[3]);
      f4raw bv = *(const f4raw*)(B + (size_t)(n0 + r) * K + k0 + kq);
      unsigned* bd = (unsigned*)(Bl + r * 72 + kq);
      bd[0] = pack2(bv[0], bv[1]);
      bd[1] = pack2(bv[2], bv[3]);
    }
    __syncthreads();
    const unsigned short* arow = Al + (wave * 16 + nl) * 72;
    v16bf a0 = load_frag_a(arow, 0, hi);
    v16bf a1 = load_frag_a(arow, 32, hi);
    v16bf b0[4], b1[4];
    for (int nt = 0; nt < 4; ++nt) b0[nt] = load_frag_b(Bl + (nt * 16 + nl) * 72, hi);
    for (int nt = 0; nt < 4; ++nt) b1[nt] = load_frag_b(Bl + (nt * 16 + nl) * 72 + 32, hi);
    for (int nt = 0; nt < 4; ++nt) acc[nt] = WMMA_BF16(a0, b0[nt], acc[nt]);
    for (int nt = 0; nt < 4; ++nt) acc[nt] = WMMA_BF16(a1, b1[nt], acc[nt]);
  }
  for (int nt = 0; nt < 4; ++nt) {
    int col = n0 + nt * 16 + nl;
    float bv = bias ? bias[col] : 0.f;
    for (int j = 0; j < 8; ++j) {
      int row = m0 + wave * 16 + j + 8 * hi;
      C[(size_t)row * N + col] = acc[nt][j] + bv;
    }
  }
}

// ---------------------------------------------------------------------------
// Stage one 64-key tile: K rows [64][128] + V-transposed rows [128][64] bf16,
// each lane moves 16B chunks; async global->LDS when available.
// ---------------------------------------------------------------------------
__device__ __forceinline__ void stage_tiles(const unsigned short* __restrict__ kg,
                                            const unsigned short* __restrict__ vtg,
                                            size_t vstride, int kpos,
                                            unsigned short* Klb, unsigned short* Vtb, int t) {
  for (int ii = t; ii < 1024; ii += 128) {          // K: 64 rows x 16 chunks
    int key = ii >> 4, c16 = ii & 15;
    const unsigned short* src = kg + key * 128 + c16 * 8;
    unsigned short* dst = Klb + key * 136 + c16 * 8;
#if HAS_ASYNC
    async_cp16(src, dst);
#else
    *(f4raw*)dst = *(const f4raw*)src;
#endif
  }
  for (int ii = t; ii < 1024; ii += 128) {          // V^T: 128 rows x 8 chunks
    int d = ii >> 3, c8 = ii & 7;
    const unsigned short* src = vtg + (size_t)d * vstride + kpos + c8 * 8;
    unsigned short* dst = Vtb + d * 72 + c8 * 8;
#if HAS_ASYNC
    async_cp16(src, dst);
#else
    *(f4raw*)dst = *(const f4raw*)src;
#endif
  }
}

// ---------------------------------------------------------------------------
// Split-K flash attention partials. grid = 392 = 2(W)*28(H)*7 chunks of 2048.
// block = 128 (4 waves); wave i owns query rows 16i..16i+15.
// ---------------------------------------------------------------------------
__global__ __launch_bounds__(128) void attn_partial(
    const float* __restrict__ qws,
    const unsigned short* __restrict__ kc0, const unsigned short* __restrict__ kc1,
    const unsigned short* __restrict__ kc2,
    const unsigned short* __restrict__ vt0, const unsigned short* __restrict__ vt1,
    const unsigned short* __restrict__ vt2,
    const float* __restrict__ cosT, const float* __restrict__ sinT,
    const int* __restrict__ loc,
    float* __restrict__ pacc, float* __restrict__ pm, float* __restrict__ plv) {
  __shared__ unsigned short Ql[64 * 136];        // RoPE'd scaled Q, bf16
  __shared__ unsigned short Kl[2][64 * 136];     // double-buffered K tile
  __shared__ unsigned short Vt[2][128 * 72];     // double-buffered V^T tile
  __shared__ unsigned short Pl[4 * 16 * 72];     // per-wave P scratch

  const int bid = blockIdx.x;
  const int w = bid / (28 * 7);
  const int rem = bid % (28 * 7);
  const int h = rem / 7;
  const int ch = rem % 7;
  int f, koff;
  const unsigned short *kc, *vtg;
  size_t Lf;
  if (ch < 4)      { f = 0; koff = ch * 2048;       kc = kc0; vtg = vt0; Lf = 8192; }
  else if (ch < 6) { f = 1; koff = (ch - 4) * 2048; kc = kc1; vtg = vt1; Lf = 4096; }
  else             { f = 2; koff = 0;               kc = kc2; vtg = vt2; Lf = 2048; }
  const int kv = h / 7;                         // GQA group size 7
  const float scale = 0.088388347648318447f;    // 1/sqrt(128)

  const int t = threadIdx.x, wave = t >> 5, lane = t & 31;
  const int nl = lane & 15, hi = lane >> 4;

  // ---- RoPE + scale Q into LDS (fragment-specific cos/sin) ----
  for (int idx = t; idx < 64 * 128; idx += 128) {
    int q = idx >> 7, d = idx & 127;
    size_t qb = (size_t)(w * 64 + q) * 3584 + (size_t)h * 128;
    float qv = qws[qb + d];
    float qr = (d < 64) ? -qws[qb + d + 64] : qws[qb + d - 64];
    int cs = ((f * 2 + w) * 64 + q) * 128 + d;
    Ql[q * 136 + d] = f2bf((qv * cosT[cs] + qr * sinT[cs]) * scale);
  }

  float m_run[8], l_run[8];
  int locrow[8];
  v8f acc[8];
  for (int j = 0; j < 8; ++j) {
    int row = wave * 16 + j + 8 * hi;
    locrow[j] = loc[(f * 2 + w) * 64 + row];
    m_run[j] = -1e30f;
    l_run[j] = 0.f;
  }
  for (int dt = 0; dt < 8; ++dt) acc[dt] = vzero8();
  __syncthreads();

  v16bf aq[4];
  {
    const unsigned short* qrow = Ql + (wave * 16 + nl) * 136;
    for (int cc = 0; cc < 4; ++cc) aq[cc] = load_frag_a(qrow, cc * 32, hi);
  }

  const unsigned short* kcb = kc + (size_t)kv * Lf * 128;
  const unsigned short* vtb = vtg + (size_t)kv * 128 * Lf;

  stage_tiles(kcb + (size_t)koff * 128, vtb, Lf, koff, Kl[0], Vt[0], t);

  for (int it = 0; it < 32; ++it) {
    const int cur = it & 1;
    wait_async();          // this wave's tile-`it` copies have landed in LDS
    __syncthreads();       // all waves' copies landed; prev compute finished
    if (it + 1 < 32) {     // prefetch next tile into alternate buffer
      int kn = koff + (it + 1) * 64;
      stage_tiles(kcb + (size_t)kn * 128, vtb, Lf, kn, Kl[1 - cur], Vt[1 - cur], t);
    }
    const unsigned short* Klc = Kl[cur];
    const unsigned short* Vtc = Vt[cur];
    const int kgb = koff + it * 64;

    // ---- S = rq * K^T : 16 queries x 64 keys per wave ----
    v8f s[4];
    for (int nt = 0; nt < 4; ++nt) {
      const unsigned short* colbase = Klc + (nt * 16 + nl) * 136;
      v16bf b[4];
      for (int cc = 0; cc < 4; ++cc) b[cc] = load_frag_b(colbase + cc * 32, hi);
      v8f sv = vzero8();
      for (int cc = 0; cc < 4; ++cc) sv = WMMA_BF16(aq[cc], b[cc], sv);
      s[nt] = sv;
    }
    // ---- mask + online softmax (row lives across 16 lanes of one VGPR) ----
    float mnew[8];
    for (int nt = 0; nt < 4; ++nt) {
      int kpcol = kgb + nt * 16 + nl;
      for (int j = 0; j < 8; ++j)
        if (kpcol > locrow[j]) s[nt][j] = -1e30f;
    }
    for (int j = 0; j < 8; ++j) {
      float mv = fmaxf(fmaxf(s[0][j], s[1][j]), fmaxf(s[2][j], s[3][j]));
      mv = fmaxf(mv, __shfl_xor(mv, 1, 32));
      mv = fmaxf(mv, __shfl_xor(mv, 2, 32));
      mv = fmaxf(mv, __shfl_xor(mv, 4, 32));
      mv = fmaxf(mv, __shfl_xor(mv, 8, 32));
      mnew[j] = fmaxf(mv, m_run[j]);
    }
    float rsum[8];
    for (int j = 0; j < 8; ++j) rsum[j] = 0.f;
    for (int nt = 0; nt < 4; ++nt)
      for (int j = 0; j < 8; ++j) {
        float p = __expf(s[nt][j] - mnew[j]);
        s[nt][j] = p;
        rsum[j] += p;
      }
    for (int j = 0; j < 8; ++j) {
      float rv = rsum[j];
      rv += __shfl_xor(rv, 1, 32);
      rv += __shfl_xor(rv, 2, 32);
      rv += __shfl_xor(rv, 4, 32);
      rv += __shfl_xor(rv, 8, 32);
      float alpha = __expf(m_run[j] - mnew[j]);
      l_run[j] = l_run[j] * alpha + rv;
      m_run[j] = mnew[j];
      for (int dt = 0; dt < 8; ++dt) acc[dt][j] *= alpha;
    }
    // ---- P via per-wave LDS scratch into A-fragment layout, then PV ----
    unsigned short* pb = Pl + wave * (16 * 72);
    for (int nt = 0; nt < 4; ++nt)
      for (int j = 0; j < 8; ++j)
        pb[(j + 8 * hi) * 72 + nt * 16 + nl] = f2bf(s[nt][j]);
    const unsigned short* prow = pb + nl * 72;
    for (int cc = 0; cc < 2; ++cc) {
      v16bf ap = load_frag_a(prow, cc * 32, hi);
      for (int dt0 = 0; dt0 < 8; dt0 += 4) {
        v16bf b[4];
        for (int e = 0; e < 4; ++e)
          b[e] = load_frag_b(Vtc + ((dt0 + e) * 16 + nl) * 72 + cc * 32, hi);
        for (int e = 0; e < 4; ++e) acc[dt0 + e] = WMMA_BF16(ap, b[e], acc[dt0 + e]);
      }
    }
  }

  // ---- unnormalized partials + stats ----
  const int pidx = (w * 28 + h) * 7 + ch;
  for (int dt = 0; dt < 8; ++dt)
    for (int j = 0; j < 8; ++j) {
      int row = wave * 16 + j + 8 * hi;
      pacc[(size_t)(pidx * 64 + row) * 128 + dt * 16 + nl] = acc[dt][j];
    }
  if (nl == 0)
    for (int j = 0; j < 8; ++j) {
      int row = wave * 16 + j + 8 * hi;
      pm[pidx * 64 + row] = m_run[j];
      plv[pidx * 64 + row] = l_run[j];
    }
}

// ---------------------------------------------------------------------------
// Merge 7 split-K partials per (w,h) -> [W*QN, H*D] for the out-projection.
// ---------------------------------------------------------------------------
__global__ __launch_bounds__(256) void attn_combine(const float* __restrict__ pacc,
                                                    const float* __restrict__ pm,
                                                    const float* __restrict__ plv,
                                                    float* __restrict__ attn) {
  int flat = blockIdx.x * 256 + threadIdx.x;
  if (flat >= 2 * 28 * 64 * 128) return;
  int d = flat & 127;
  int q = (flat >> 7) & 63;
  int hw = flat >> 13;           // w*28 + h
  int h = hw % 28, w = hw / 28;
  int pbase = hw * 7;
  float M = -1e30f;
  for (int c = 0; c < 7; ++c) M = fmaxf(M, pm[(pbase + c) * 64 + q]);
  float L = 0.f, val = 0.f;
  for (int c = 0; c < 7; ++c) {
    float e = __expf(pm[(pbase + c) * 64 + q] - M);
    L += plv[(pbase + c) * 64 + q] * e;
    val += pacc[(size_t)((pbase + c) * 64 + q) * 128 + d] * e;
  }
  attn[(size_t)(w * 64 + q) * 3584 + h * 128 + d] = val / L;
}

// ---------------------------------------------------------------------------
extern "C" void kernel_launch(void* const* d_in, const int* in_sizes, int n_in,
                              void* d_out, int out_size, void* d_ws, size_t ws_size,
                              hipStream_t stream) {
  (void)in_sizes; (void)n_in; (void)out_size; (void)ws_size;
  const float* hidden = (const float*)d_in[0];
  const float* wq     = (const float*)d_in[1];
  const float* bq     = (const float*)d_in[2];
  const float* wo     = (const float*)d_in[3];
  const float* k0     = (const float*)d_in[4];
  const float* v0     = (const float*)d_in[5];
  const float* k1     = (const float*)d_in[6];
  const float* v1     = (const float*)d_in[7];
  const float* k2     = (const float*)d_in[8];
  const float* v2     = (const float*)d_in[9];
  const int*   loc    = (const int*)d_in[10];
  const float* cosT   = (const float*)d_in[11];
  const float* sinT   = (const float*)d_in[12];
  float* out = (float*)d_out;

  float* ws      = (float*)d_ws;
  float* q_ws    = ws;                       // 128*3584   = 458752
  float* attn_ws = q_ws + 458752;            // 458752
  float* pacc    = attn_ws + 458752;         // 392*64*128 = 3211264
  float* pm      = pacc + 3211264;           // 25088
  float* plv     = pm + 25088;               // 25088
  unsigned short* us  = (unsigned short*)(plv + 25088);
  unsigned short* kc0 = us;                  // 4*8192*128 = 4194304
  unsigned short* kc1 = kc0 + 4194304;       // 2097152
  unsigned short* kc2 = kc1 + 2097152;       // 1048576
  unsigned short* vt0 = kc2 + 1048576;       // 4194304
  unsigned short* vt1 = vt0 + 4194304;       // 2097152
  unsigned short* vt2 = vt1 + 2097152;       // 1048576

  // 0) bf16 K/V caches (L2-resident: 29 MB total vs 192 MB L2); V transposed
  convert_flat<<<4096, 256, 0, stream>>>(k0, kc0, 1048576);
  convert_flat<<<2048, 256, 0, stream>>>(k1, kc1, 524288);
  convert_flat<<<1024, 256, 0, stream>>>(k2, kc2, 262144);
  transpose_v <<<4096, 256, 0, stream>>>(v0, vt0, 8192);
  transpose_v <<<2048, 256, 0, stream>>>(v1, vt1, 4096);
  transpose_v <<<1024, 256, 0, stream>>>(v2, vt2, 2048);
  // 1) Q = hidden @ wq^T + bq
  gemm_bf16<<<dim3(2, 56), 128, 0, stream>>>(hidden, wq, bq, q_ws, 128, 3584, 3584);
  // 2) split-K flash attention partials (RoPE fused, async double-buffered)
  attn_partial<<<392, 128, 0, stream>>>(q_ws, kc0, kc1, kc2, vt0, vt1, vt2,
                                        cosT, sinT, loc, pacc, pm, plv);
  // 3) merge partials
  attn_combine<<<1792, 256, 0, stream>>>(pacc, pm, plv, attn_ws);
  // 4) out = attn @ wo^T
  gemm_bf16<<<dim3(2, 56), 128, 0, stream>>>(attn_ws, wo, nullptr, out, 128, 3584, 3584);
}
